// EncoderBlock_773094113518
// MI455X (gfx1250) — compile-verified, matrix operand order checked by reading, not run on hardware
//
#include <hip/hip_runtime.h>
#include <math.h>

// ---------- types ----------
typedef __attribute__((ext_vector_type(16))) _Float16 v16h;
typedef __attribute__((ext_vector_type(8)))  float    v8f;
typedef unsigned int u32x4 __attribute__((ext_vector_type(4)));
typedef float        f32x4 __attribute__((ext_vector_type(4)));

#define DEV static __device__ __forceinline__

// Problem constants (match reference)
constexpr int Bc  = 2;
constexpr int S   = 2048;
constexpr int D   = 1024;
constexpr int H   = 16;
constexpr int DH  = 64;
constexpr int NB  = 16;       // S / 128
constexpr int TOK = Bc * S;   // 4096 rows

// Load a 16-half WMMA A/B fragment: lane holds K=[kb..kb+7] in v0..3 and
// K=[kb+16..kb+23] in v4..7 (kb = (lane>=16)*8 added by caller), both 16B reads.
DEV v16h load_frag(const _Float16* p) {
  union { u32x4 u[2]; v16h h; } f;
  f.u[0] = *reinterpret_cast<const u32x4*>(p);
  f.u[1] = *reinterpret_cast<const u32x4*>(p + 16);
  return f.h;
}

DEV v8f wmma_f16(v16h a, v16h b, v8f c) {
  return __builtin_amdgcn_wmma_f32_16x16x32_f16(false, a, false, b, (short)0, c,
                                                false, false);
}

DEV v8f zero8() {
  v8f z;
#pragma unroll
  for (int i = 0; i < 8; ++i) z[i] = 0.0f;
  return z;
}

DEV unsigned int pack2h(float a, float b) {
  union { _Float16 h[2]; unsigned int u; } p;
  p.h[0] = (_Float16)a;
  p.h[1] = (_Float16)b;
  return p.u;
}

// ---------- LayerNorm: one 1024-wide row per 256-thread block ----------
__global__ __launch_bounds__(256) void ln_kernel(const float* __restrict__ x,
                                                 const float* __restrict__ g,
                                                 const float* __restrict__ be,
                                                 float* __restrict__ out) {
  __shared__ float rs[256], rq[256];
  const int row = blockIdx.x, tid = threadIdx.x;
  const float* xp = x + (size_t)row * D + tid * 4;
  f32x4 f = *reinterpret_cast<const f32x4*>(xp);
  rs[tid] = f.x + f.y + f.z + f.w;
  rq[tid] = f.x * f.x + f.y * f.y + f.z * f.z + f.w * f.w;
  __syncthreads();
  for (int off = 128; off > 0; off >>= 1) {
    if (tid < off) { rs[tid] += rs[tid + off]; rq[tid] += rq[tid + off]; }
    __syncthreads();
  }
  const float mean = rs[0] * (1.0f / D);
  const float var  = rq[0] * (1.0f / D) - mean * mean;
  const float inv  = rsqrtf(var + 1e-5f);
  float* op = out + (size_t)row * D + tid * 4;
  const float* gp = g + tid * 4;
  const float* bp = be + tid * 4;
  op[0] = (f.x - mean) * inv * gp[0] + bp[0];
  op[1] = (f.y - mean) * inv * gp[1] + bp[1];
  op[2] = (f.z - mean) * inv * gp[2] + bp[2];
  op[3] = (f.w - mean) * inv * gp[3] + bp[3];
}

// ---------- Generic 128x128-tile WMMA GEMM, software-pipelined ----------
enum { EPI_QKV = 0, EPI_BIAS_RES_F32 = 1, EPI_BIAS_GELU_F16 = 2 };

template <bool AHALF, int EPI>
__global__ __launch_bounds__(256) void gemm128(const void*  __restrict__ Aptr,
                                               const float* __restrict__ Bw,
                                               const float* __restrict__ bias,
                                               const float* __restrict__ resid,
                                               void* __restrict__ outp,
                                               int M, int N, int K) {
  __shared__ __align__(16) _Float16 Al[128 * 32];   // A tile, row-major [128][32]
  __shared__ __align__(16) _Float16 Bt[128 * 32];   // B tile transposed [n][k]
  const int tid  = threadIdx.x;
  const int n0   = blockIdx.x * 128;
  const int m0   = blockIdx.y * 128;
  const int lane = tid & 31;
  const int l15  = lane & 15;
  const int hi   = lane >> 4;
  const int kb   = hi * 8;
  const int wave = tid >> 5;            // 8 waves: 4 along M x 2 along N
  const int wm   = wave >> 1, wn = wave & 1;

  // per-thread staging coordinates
  const int rowA = tid >> 1, c0A = (tid & 1) * 16;   // A: 128x32, 16 elems/thread
  const int krB  = tid >> 3, c0B = (tid & 7) * 16;   // B: 32x128, 16 elems/thread

  const float*    AgF = (const float*)Aptr    + (size_t)(m0 + rowA) * K + c0A;
  const _Float16* AgH = (const _Float16*)Aptr + (size_t)(m0 + rowA) * K + c0A;
  const float*    Bg  = Bw + (size_t)krB * N + n0 + c0B;

  v8f acc[2][4];
#pragma unroll
  for (int i = 0; i < 2; ++i)
#pragma unroll
    for (int j = 0; j < 4; ++j) acc[i][j] = zero8();

  // staging registers for the current chunk
  f32x4 fa[4];
  u32x4 ua[2];
  f32x4 fb[4];

  // preload chunk 0
  if (AHALF) {
    ua[0] = reinterpret_cast<const u32x4*>(AgH)[0];
    ua[1] = reinterpret_cast<const u32x4*>(AgH)[1];
  } else {
#pragma unroll
    for (int t = 0; t < 4; ++t) fa[t] = reinterpret_cast<const f32x4*>(AgF)[t];
  }
#pragma unroll
  for (int t = 0; t < 4; ++t) fb[t] = reinterpret_cast<const f32x4*>(Bg)[t];

  for (int k0 = 0; k0 < K; k0 += 32) {
    // ---- stage current chunk into LDS ----
    if (AHALF) {
      reinterpret_cast<u32x4*>(&Al[rowA * 32 + c0A])[0] = ua[0];
      reinterpret_cast<u32x4*>(&Al[rowA * 32 + c0A])[1] = ua[1];
    } else {
      u32x4 w0, w1;
      w0.x = pack2h(fa[0].x, fa[0].y); w0.y = pack2h(fa[0].z, fa[0].w);
      w0.z = pack2h(fa[1].x, fa[1].y); w0.w = pack2h(fa[1].z, fa[1].w);
      w1.x = pack2h(fa[2].x, fa[2].y); w1.y = pack2h(fa[2].z, fa[2].w);
      w1.z = pack2h(fa[3].x, fa[3].y); w1.w = pack2h(fa[3].z, fa[3].w);
      reinterpret_cast<u32x4*>(&Al[rowA * 32 + c0A])[0] = w0;
      reinterpret_cast<u32x4*>(&Al[rowA * 32 + c0A])[1] = w1;
    }
#pragma unroll
    for (int t = 0; t < 4; ++t) {   // transposed scatter (unavoidable b16 stores)
      Bt[(c0B + 4 * t + 0) * 32 + krB] = (_Float16)fb[t].x;
      Bt[(c0B + 4 * t + 1) * 32 + krB] = (_Float16)fb[t].y;
      Bt[(c0B + 4 * t + 2) * 32 + krB] = (_Float16)fb[t].z;
      Bt[(c0B + 4 * t + 3) * 32 + krB] = (_Float16)fb[t].w;
    }
    __syncthreads();

    // ---- prefetch next chunk into registers (overlaps with WMMAs below) ----
    if (k0 + 32 < K) {
      if (AHALF) {
        ua[0] = reinterpret_cast<const u32x4*>(AgH + k0 + 32)[0];
        ua[1] = reinterpret_cast<const u32x4*>(AgH + k0 + 32)[1];
      } else {
#pragma unroll
        for (int t = 0; t < 4; ++t)
          fa[t] = reinterpret_cast<const f32x4*>(AgF + k0 + 32)[t];
      }
#pragma unroll
      for (int t = 0; t < 4; ++t)
        fb[t] = reinterpret_cast<const f32x4*>(Bg + (size_t)(k0 + 32) * N)[t];
      if (k0 + 64 < K) {  // pull chunk k0+64 lines toward the WGP early
        __builtin_prefetch(Bg + (size_t)(k0 + 64) * N, 0, 0);
        if (AHALF) __builtin_prefetch(AgH + k0 + 64, 0, 0);
        else       __builtin_prefetch(AgF + k0 + 64, 0, 0);
      }
    }

    // ---- WMMAs on current chunk ----
    v16h af[2];
#pragma unroll
    for (int mt = 0; mt < 2; ++mt)
      af[mt] = load_frag(&Al[(wm * 32 + mt * 16 + l15) * 32 + kb]);
#pragma unroll
    for (int nt = 0; nt < 4; ++nt) {
      v16h bf = load_frag(&Bt[(wn * 64 + nt * 16 + l15) * 32 + kb]);
#pragma unroll
      for (int mt = 0; mt < 2; ++mt) acc[mt][nt] = wmma_f16(af[mt], bf, acc[mt][nt]);
    }
    __syncthreads();
  }

  // Epilogue. C layout: vgpr e -> row (e or e+8 per half-wave), col = l15.
#pragma unroll
  for (int mt = 0; mt < 2; ++mt)
#pragma unroll
    for (int nt = 0; nt < 4; ++nt)
#pragma unroll
      for (int e = 0; e < 8; ++e) {
        const int m = m0 + wm * 32 + mt * 16 + e + 8 * hi;
        const int n = n0 + wn * 64 + nt * 16 + l15;
        float vv = acc[mt][nt][e] + bias[n];
        if (EPI == EPI_QKV) {  // scatter to [B,H,S,DH] as f16
          const int b = m >> 11, s = m & 2047, hh = n >> 6, dh = n & 63;
          ((_Float16*)outp)[((size_t)(b * H + hh) * S + s) * DH + dh] = (_Float16)vv;
        } else if (EPI == EPI_BIAS_RES_F32) {
          const size_t idx = (size_t)m * N + n;
          ((float*)outp)[idx] = vv + resid[idx];
        } else {  // exact GeLU -> f16
          const float ge = 0.5f * vv * (1.0f + erff(vv * 0.70710678118654752f));
          ((_Float16*)outp)[(size_t)m * N + n] = (_Float16)ge;
        }
      }
}

// ---------- Block-sparse flash attention ----------
// grid (NB, H, B); 256 threads = 8 waves; wave owns 16 query rows.
__global__ __launch_bounds__(256) void attn_kernel(const _Float16* __restrict__ q,
                                                   const _Float16* __restrict__ kk_,
                                                   const _Float16* __restrict__ vv_,
                                                   const int* __restrict__ layout,
                                                   float* __restrict__ ctx) {
  __shared__ __align__(16) _Float16 VT[64 * 128];       // V^T [dh][key]
  __shared__ __align__(16) _Float16 Ps[8 * 16 * 128];   // per-wave P [16][128]
  const int tid  = threadIdx.x;
  const int i    = blockIdx.x;   // query block
  const int h    = blockIdx.y;
  const int b    = blockIdx.z;
  const int bh   = b * H + h;
  const int lane = tid & 31, l15 = lane & 15, hi = lane >> 4;
  const int kb   = hi * 8;
  const int wave = tid >> 5;

  const size_t headBase = (size_t)bh * S * DH;

  // Q fragments for this wave's rows, straight from global (K-contig layout)
  const _Float16* qrow = q + headBase + (size_t)(i * 128 + wave * 16 + l15) * DH;
  const v16h aQ0 = load_frag(qrow + 0  + kb);
  const v16h aQ1 = load_frag(qrow + 32 + kb);

  float m_run[8], l_run[8];
  v8f o[4];
#pragma unroll
  for (int r = 0; r < 8; ++r) { m_run[r] = -1e30f; l_run[r] = 0.0f; }
#pragma unroll
  for (int t = 0; t < 4; ++t) o[t] = zero8();

  const float scale = 0.125f;  // DH^-0.5

  for (int j = 0; j <= i; ++j) {
    if (!layout[h * NB * NB + i * NB + j]) continue;  // wave-uniform

    { // load V block j into LDS transposed
      const int key = tid >> 1, d0 = (tid & 1) * 32;
      const _Float16* vg = vv_ + headBase + (size_t)(j * 128 + key) * DH + d0;
#pragma unroll
      for (int t = 0; t < 32; ++t) VT[(d0 + t) * 128 + key] = vg[t];
    }
    __syncthreads();

    // scores S = Q K^T (K fragments straight from global)
    float sc[8][8];
#pragma unroll
    for (int nt = 0; nt < 8; ++nt) {
      const _Float16* krow = kk_ + headBase + (size_t)(j * 128 + nt * 16 + l15) * DH;
      v16h b0 = load_frag(krow + 0  + kb);
      v16h b1 = load_frag(krow + 32 + kb);
      v8f cs = zero8();
      cs = wmma_f16(aQ0, b0, cs);
      cs = wmma_f16(aQ1, b1, cs);
#pragma unroll
      for (int e = 0; e < 8; ++e) {
        float s = cs[e] * scale;
        if (j == i) {  // intra-block causal
          const int col = nt * 16 + l15;
          const int row = wave * 16 + e + 8 * hi;
          if (col > row) s = -1e9f;
        }
        sc[nt][e] = s;
      }
    }

    // online softmax per row (row values live in one 16-lane half)
#pragma unroll
    for (int r = 0; r < 8; ++r) {
      float mv = sc[0][r];
#pragma unroll
      for (int nt = 1; nt < 8; ++nt) mv = fmaxf(mv, sc[nt][r]);
      mv = fmaxf(mv, __shfl_xor(mv, 8, 32));
      mv = fmaxf(mv, __shfl_xor(mv, 4, 32));
      mv = fmaxf(mv, __shfl_xor(mv, 2, 32));
      mv = fmaxf(mv, __shfl_xor(mv, 1, 32));
      const float mnew = fmaxf(m_run[r], mv);
      const float corr = __expf(m_run[r] - mnew);
      float rsum = 0.0f;
#pragma unroll
      for (int nt = 0; nt < 8; ++nt) {
        const float p = __expf(sc[nt][r] - mnew);
        sc[nt][r] = p;
        rsum += p;
      }
      rsum += __shfl_xor(rsum, 8, 32);
      rsum += __shfl_xor(rsum, 4, 32);
      rsum += __shfl_xor(rsum, 2, 32);
      rsum += __shfl_xor(rsum, 1, 32);
      l_run[r] = l_run[r] * corr + rsum;
      m_run[r] = mnew;
#pragma unroll
      for (int t = 0; t < 4; ++t) o[t][r] *= corr;
    }

    // P: C-layout regs -> LDS (A-layout source for next WMMA)
    _Float16* pw = &Ps[wave * 16 * 128];
#pragma unroll
    for (int nt = 0; nt < 8; ++nt)
#pragma unroll
      for (int r = 0; r < 8; ++r)
        pw[(r + 8 * hi) * 128 + nt * 16 + l15] = (_Float16)sc[nt][r];
    __syncthreads();

    // O += P x V
#pragma unroll
    for (int kc = 0; kc < 4; ++kc) {
      v16h aP = load_frag(&pw[l15 * 128 + kc * 32 + kb]);
#pragma unroll
      for (int nt2 = 0; nt2 < 4; ++nt2) {
        v16h bV = load_frag(&VT[(nt2 * 16 + l15) * 128 + kc * 32 + kb]);
        o[nt2] = wmma_f16(aP, bV, o[nt2]);
      }
    }
    __syncthreads();
  }

  // normalize and write ctx in token-major [TOK, D] fp32
#pragma unroll
  for (int r = 0; r < 8; ++r) {
    const float inv = 1.0f / l_run[r];
    const int tok = b * S + i * 128 + wave * 16 + r + 8 * hi;
#pragma unroll
    for (int nt2 = 0; nt2 < 4; ++nt2) {
      const int col = h * DH + nt2 * 16 + l15;
      ctx[(size_t)tok * D + col] = o[nt2][r] * inv;
    }
  }
}

// ---------- launch ----------
extern "C" void kernel_launch(void* const* d_in, const int* in_sizes, int n_in,
                              void* d_out, int out_size, void* d_ws, size_t ws_size,
                              hipStream_t stream) {
  (void)in_sizes; (void)n_in; (void)out_size; (void)ws_size;
  const float* x   = (const float*)d_in[0];
  const float* l1g = (const float*)d_in[1];
  const float* l1b = (const float*)d_in[2];
  const float* Wq  = (const float*)d_in[3];
  const float* bq  = (const float*)d_in[4];
  const float* Wk  = (const float*)d_in[5];
  const float* bk  = (const float*)d_in[6];
  const float* Wv  = (const float*)d_in[7];
  const float* bv  = (const float*)d_in[8];
  const float* Wo  = (const float*)d_in[9];
  const float* bo  = (const float*)d_in[10];
  const float* l2g = (const float*)d_in[11];
  const float* l2b = (const float*)d_in[12];
  const float* W1  = (const float*)d_in[13];
  const float* b1  = (const float*)d_in[14];
  const float* W2  = (const float*)d_in[15];
  const float* b2  = (const float*)d_in[16];
  const int* layout = (const int*)d_in[17];

  // workspace overlay (peak 72 MB):
  //  [0,16M)  h / h2 fp32        [16M,24M) q16  [24M,32M) k16  [32M,40M) v16
  //  [40M,56M) ctx fp32          [56M,72M) xr fp32
  //  [16M,48M) ff f16 (reuses qkv+ctx head after they are dead)
  char* w = (char*)d_ws;
  float*    hbuf = (float*)(w);
  _Float16* q16  = (_Float16*)(w + (size_t)(16u << 20));
  _Float16* k16  = (_Float16*)(w + (size_t)(24u << 20));
  _Float16* v16  = (_Float16*)(w + (size_t)(32u << 20));
  float*    ctx  = (float*)(w + (size_t)(40u << 20));
  float*    xr   = (float*)(w + (size_t)(56u << 20));
  _Float16* ff   = (_Float16*)(w + (size_t)(16u << 20));

  const dim3 blk(256);

  // LN1
  ln_kernel<<<TOK, blk, 0, stream>>>(x, l1g, l1b, hbuf);
  // QKV projections -> f16 [B,H,S,DH]
  gemm128<false, EPI_QKV><<<dim3(D / 128, TOK / 128), blk, 0, stream>>>(
      hbuf, Wq, bq, nullptr, q16, TOK, D, D);
  gemm128<false, EPI_QKV><<<dim3(D / 128, TOK / 128), blk, 0, stream>>>(
      hbuf, Wk, bk, nullptr, k16, TOK, D, D);
  gemm128<false, EPI_QKV><<<dim3(D / 128, TOK / 128), blk, 0, stream>>>(
      hbuf, Wv, bv, nullptr, v16, TOK, D, D);
  // block-sparse attention
  attn_kernel<<<dim3(NB, H, Bc), blk, 0, stream>>>(q16, k16, v16, layout, ctx);
  // out projection + residual -> xr
  gemm128<false, EPI_BIAS_RES_F32><<<dim3(D / 128, TOK / 128), blk, 0, stream>>>(
      ctx, Wo, bo, x, xr, TOK, D, D);
  // LN2 (reuse hbuf)
  ln_kernel<<<TOK, blk, 0, stream>>>(xr, l2g, l2b, hbuf);
  // MLP1 + GeLU -> f16 ff
  gemm128<false, EPI_BIAS_GELU_F16><<<dim3(4 * D / 128, TOK / 128), blk, 0, stream>>>(
      hbuf, W1, b1, nullptr, ff, TOK, 4 * D, D);
  // MLP2 + residual -> d_out (fp32)
  gemm128<true, EPI_BIAS_RES_F32><<<dim3(D / 128, TOK / 128), blk, 0, stream>>>(
      ff, W2, b2, xr, d_out, TOK, D, 4 * D);
}
